// CGGRUNet_85993835200797
// MI455X (gfx1250) — matrix operand order repaired and checked by reading, not compiled, not run on hardware
//
#include <hip/hip_runtime.h>
#include <hip/hip_bf16.h>

#define N_NODES 10000
#define N_EDGES 100000
#define NB      64
#define DIM     64
#define NF      92
#define EF      50
#define HH      128
#define EPB     128   // edges per block (conv + nn1)

#define USE_ASYNC_LDS 1

typedef _Float16 h16 __attribute__((ext_vector_type(16)));
typedef _Float16 h8  __attribute__((ext_vector_type(8)));
typedef float    v8f __attribute__((ext_vector_type(8)));

union H16U { h16 v; h8 g[2]; };

#define WMMA_F16(A, B, C) \
  __builtin_amdgcn_wmma_f32_16x16x32_f16(false, (A), false, (B), (short)0, (C), false, false)

// ---------------- helpers ----------------
__device__ __forceinline__ unsigned fkey(float f) {
  unsigned u = __float_as_uint(f);
  return (u & 0x80000000u) ? ~u : (u | 0x80000000u);
}
__device__ __forceinline__ float keyf(unsigned k) {
  unsigned u = (k & 0x80000000u) ? (k & 0x7fffffffu) : ~k;
  return __uint_as_float(u);
}
__device__ __forceinline__ float sigm(float x) { return 1.f / (1.f + __expf(-x)); }

__global__ void zero_f(float* p, int n) {
  int i = blockIdx.x * blockDim.x + threadIdx.x;
  if (i < n) p[i] = 0.f;
}
__global__ void zero_u(unsigned* p, int n) {
  int i = blockIdx.x * blockDim.x + threadIdx.x;
  if (i < n) p[i] = 0u;
}

// ---------------- weight reorder (one-time per call, tiny) ----------------
// W2T[o][k], k = i*128 + h, value = nn2_w[(i*64+o)*128 + h]  (f16, 1MB, L2-resident)
__global__ void prep_w2t(const float* __restrict__ nn2_w, _Float16* __restrict__ w2t) {
  int idx = blockIdx.x * blockDim.x + threadIdx.x;
  if (idx >= 64 * 8192) return;
  int o = idx / 8192, k = idx % 8192;
  int i = k / HH, h = k % HH;
  w2t[idx] = (_Float16)nn2_w[(i * DIM + o) * HH + h];
}
// bbT[o][i] = nn2_b[i*64+o]  (bias fold-in, f16)
__global__ void prep_bbt(const float* __restrict__ nn2_b, _Float16* __restrict__ bbt) {
  int idx = blockIdx.x * blockDim.x + threadIdx.x;
  if (idx >= DIM * DIM) return;
  int o = idx / DIM, i = idx % DIM;
  bbt[idx] = (_Float16)nn2_b[i * DIM + o];
}
// n1bt[n][k] = nn1_w[n][k] zero-padded K 50->64 (f16)
__global__ void prep_n1bt(const float* __restrict__ nn1_w, _Float16* __restrict__ bt) {
  int idx = blockIdx.x * blockDim.x + threadIdx.x;
  if (idx >= HH * 64) return;
  int n = idx / 64, k = idx % 64;
  bt[idx] = (k < EF) ? (_Float16)nn1_w[n * EF + k] : (_Float16)0.f;
}

// ---------------- lin0: out = relu(x @ W.T + b), also f16 copy ----------------
__global__ void lin0_kernel(const float* __restrict__ x, const float* __restrict__ w,
                            const float* __restrict__ b, float* __restrict__ out,
                            _Float16* __restrict__ xs) {
  __shared__ float xr[4][NF];
  int ln = threadIdx.x / DIM, d = threadIdx.x % DIM;
  int n = blockIdx.x * 4 + ln;
  for (int j = d; j < NF; j += DIM) xr[ln][j] = (n < N_NODES) ? x[n * NF + j] : 0.f;
  __syncthreads();
  if (n >= N_NODES) return;
  float s = b[d];
  for (int j = 0; j < NF; ++j) s += xr[ln][j] * w[d * NF + j];
  s = fmaxf(s, 0.f);
  out[n * DIM + d] = s;
  xs[n * DIM + d] = (_Float16)s;
}

// ---------------- nn1: he = relu(edge_attr @ nn1_w.T + b), f16, via WMMA ----------------
__global__ __launch_bounds__(256) void nn1_wmma(const float* __restrict__ edge_attr,
                                                const float* __restrict__ nn1_b,
                                                const _Float16* __restrict__ bt,
                                                _Float16* __restrict__ he) {
  __shared__ _Float16 ea[128][64];
  int tid = threadIdx.x, lane = tid & 31, wave = tid >> 5;
  int eb = blockIdx.x * 128;
  bool full = (eb + 128 <= N_EDGES);
  for (int idx = tid; idx < 128 * 64; idx += 256) {
    int r = idx >> 6, k = idx & 63;
    int e = eb + r;
    ea[r][k] = (e < N_EDGES && k < EF) ? (_Float16)edge_attr[e * EF + k] : (_Float16)0.f;
  }
  __syncthreads();
  int r0 = wave * 16;
  int m = lane & 15, hi = lane >> 4;
  int aoff0 = hi ? 16 : 0, aoff1 = (hi ? 16 : 0) + 8;
  int boff = hi ? 16 : 0;
  int e0 = eb + r0;
  for (int nt = 0; nt < 8; ++nt) {
    v8f acc = {0.f, 0.f, 0.f, 0.f, 0.f, 0.f, 0.f, 0.f};
    int n = nt * 16 + m;
#pragma unroll
    for (int kc = 0; kc < 2; ++kc) {
      H16U a;
      const _Float16* arow = &ea[r0 + m][kc * 32];
      a.g[0] = *(const h8*)(arow + aoff0);
      a.g[1] = *(const h8*)(arow + aoff1);
      h16 bf = *(const h16*)(bt + n * 64 + kc * 32 + boff);
      acc = WMMA_F16(a.v, bf, acc);
    }
    float bias = nn1_b[n];
    if (full) {
#pragma unroll
      for (int rr = 0; rr < 8; ++rr)
        he[(e0 + rr + hi * 8) * HH + n] = (_Float16)fmaxf(acc[rr] + bias, 0.f);
    } else {
#pragma unroll
      for (int rr = 0; rr < 8; ++rr) {
        int e = e0 + rr + hi * 8;
        if (e < N_EDGES) he[e * HH + n] = (_Float16)fmaxf(acc[rr] + bias, 0.f);
      }
    }
  }
}

// ---------------- degree ----------------
__global__ void deg_kernel(const int* __restrict__ ei, float* __restrict__ deg) {
  int e = blockIdx.x * blockDim.x + threadIdx.x;
  if (e < N_EDGES) atomicAdd(&deg[ei[N_EDGES + e]], 1.f);
}
__global__ void invdeg_kernel(const float* __restrict__ deg, float* __restrict__ inv) {
  int n = blockIdx.x * blockDim.x + threadIdx.x;
  if (n < N_NODES) inv[n] = (deg[n] > 0.f) ? 1.f / fmaxf(deg[n], 1.f) : 0.f;
}

// ---------------- conv message pass: fused Z[E,8192] @ W2T, scatter-add to agg -------
// Z[e, i*128+h] = xs[src[e],i]*he[e,h] built on the fly; 32 edges per wave (two 16-row
// A sub-tiles share every B fragment -> halves L2 traffic, 8 WMMA per 4 B loads).
__global__ __launch_bounds__(128) void conv_msg(const int* __restrict__ ei,
                                                const _Float16* __restrict__ xs,
                                                const _Float16* __restrict__ he,
                                                const _Float16* __restrict__ w2t,
                                                const _Float16* __restrict__ bbt,
                                                float* __restrict__ agg) {
  __shared__ _Float16 hel[EPB][HH];  // 32KB
  __shared__ _Float16 xsl[EPB][DIM]; // 16KB
  __shared__ int srcl[EPB];
  __shared__ int dstl[EPB];
  int tid = threadIdx.x, lane = tid & 31, wave = tid >> 5;
  int eb = blockIdx.x * EPB;
  bool full = (eb + EPB <= N_EDGES);
  for (int r = tid; r < EPB; r += 128) {
    int e = eb + r;
    srcl[r] = (e < N_EDGES) ? ei[e] : 0;
    dstl[r] = (e < N_EDGES) ? ei[N_EDGES + e] : 0;
  }
  __syncthreads();
#if USE_ASYNC_LDS
  if (full) {
    // he: 32KB contiguous -> async copy to LDS (16B per lane per issue)
    unsigned long hg = (unsigned long)(he + (unsigned long)eb * HH) + (unsigned long)tid * 16;
    unsigned hl = (unsigned)(unsigned long)(&hel[0][0]) + tid * 16;
#pragma unroll
    for (int it = 0; it < (EPB * HH * 2) / (128 * 16); ++it) {  // 16 issues/thread
      asm volatile("global_load_async_to_lds_b128 %0, %1, off" ::"v"(hl), "v"(hg)
                   : "memory");
      hg += 128 * 16;
      hl += 128 * 16;
    }
    // xs gather: per-row 128B = 8 x 16B chunks
    unsigned xbase = (unsigned)(unsigned long)(&xsl[0][0]);
#pragma unroll
    for (int c = 0; c < 8; ++c) {
      int idx = c * 128 + tid;
      int row = idx >> 3, part = idx & 7;
      unsigned long g = (unsigned long)(xs + (unsigned long)srcl[row] * DIM) + part * 16;
      unsigned l = xbase + row * 128 + part * 16;
      asm volatile("global_load_async_to_lds_b128 %0, %1, off" ::"v"(l), "v"(g)
                   : "memory");
    }
    asm volatile("s_wait_asynccnt 0" ::: "memory");
  } else
#endif
  {
    const unsigned* heg = (const unsigned*)he;
    unsigned* helu = (unsigned*)&hel[0][0];
    for (int idx = tid; idx < EPB * 64; idx += 128) {
      int r = idx >> 6, e = eb + r;
      helu[idx] = (e < N_EDGES) ? heg[e * 64 + (idx & 63)] : 0u;
    }
    const unsigned* xg = (const unsigned*)xs;
    unsigned* xlu = (unsigned*)&xsl[0][0];
    for (int idx = tid; idx < EPB * 32; idx += 128) {
      int r = idx >> 5, e = eb + r;
      xlu[idx] = (e < N_EDGES) ? xg[srcl[r] * 32 + (idx & 31)] : 0u;
    }
  }
  __syncthreads();

  int r0 = wave * 32;
  if (eb + r0 >= N_EDGES) return;  // no barriers past this point
  int m = lane & 15, hi = lane >> 4;
  int aoff0 = hi ? 16 : 0, aoff1 = (hi ? 16 : 0) + 8;
  int boff = hi ? 16 : 0;
  const _Float16* herA = &hel[r0 + m][0];
  const _Float16* herB = &hel[r0 + 16 + m][0];
  const _Float16* xsA = &xsl[r0 + m][0];
  const _Float16* xsB = &xsl[r0 + 16 + m][0];
  v8f za = {0.f, 0.f, 0.f, 0.f, 0.f, 0.f, 0.f, 0.f};
  v8f a0 = za, a1 = za, a2 = za, a3 = za;
  v8f b0 = za, b1 = za, b2 = za, b3 = za;

  for (int i = 0; i < DIM; ++i) {
    _Float16 sxA = xsA[i], sxB = xsB[i];
#pragma unroll
    for (int hc = 0; hc < 4; ++hc) {
      H16U fa, fb;
      const _Float16* hpA = herA + hc * 32;
      const _Float16* hpB = herB + hc * 32;
      fa.g[0] = *(const h8*)(hpA + aoff0);
      fa.g[1] = *(const h8*)(hpA + aoff1);
      fb.g[0] = *(const h8*)(hpB + aoff0);
      fb.g[1] = *(const h8*)(hpB + aoff1);
      fa.v = fa.v * sxA;  // Z chunk = xs[:,i] (x) he[:,hc*32..+32)
      fb.v = fb.v * sxB;
      int kg = (i * 4 + hc) * 32 + boff;
      h16 wf0 = *(const h16*)(w2t + (0 * 16 + m) * 8192 + kg);
      h16 wf1 = *(const h16*)(w2t + (1 * 16 + m) * 8192 + kg);
      h16 wf2 = *(const h16*)(w2t + (2 * 16 + m) * 8192 + kg);
      h16 wf3 = *(const h16*)(w2t + (3 * 16 + m) * 8192 + kg);
      a0 = WMMA_F16(fa.v, wf0, a0);
      b0 = WMMA_F16(fb.v, wf0, b0);
      a1 = WMMA_F16(fa.v, wf1, a1);
      b1 = WMMA_F16(fb.v, wf1, b1);
      a2 = WMMA_F16(fa.v, wf2, a2);
      b2 = WMMA_F16(fb.v, wf2, b2);
      a3 = WMMA_F16(fa.v, wf3, a3);
      b3 = WMMA_F16(fb.v, wf3, b3);
    }
  }
  // fold nn2_b: msg += xs_src @ Bb  (two extra K-chunks, A = raw xs)
#pragma unroll
  for (int ic = 0; ic < 2; ++ic) {
    H16U fa, fb;
    fa.g[0] = *(const h8*)(xsA + ic * 32 + aoff0);
    fa.g[1] = *(const h8*)(xsA + ic * 32 + aoff1);
    fb.g[0] = *(const h8*)(xsB + ic * 32 + aoff0);
    fb.g[1] = *(const h8*)(xsB + ic * 32 + aoff1);
    int kb = ic * 32 + boff;
    h16 wf0 = *(const h16*)(bbt + (0 * 16 + m) * 64 + kb);
    h16 wf1 = *(const h16*)(bbt + (1 * 16 + m) * 64 + kb);
    h16 wf2 = *(const h16*)(bbt + (2 * 16 + m) * 64 + kb);
    h16 wf3 = *(const h16*)(bbt + (3 * 16 + m) * 64 + kb);
    a0 = WMMA_F16(fa.v, wf0, a0);
    b0 = WMMA_F16(fb.v, wf0, b0);
    a1 = WMMA_F16(fa.v, wf1, a1);
    b1 = WMMA_F16(fb.v, wf1, b1);
    a2 = WMMA_F16(fa.v, wf2, a2);
    b2 = WMMA_F16(fb.v, wf2, b2);
    a3 = WMMA_F16(fa.v, wf3, a3);
    b3 = WMMA_F16(fb.v, wf3, b3);
  }
  if (full) {
#pragma unroll
    for (int rr = 0; rr < 8; ++rr) {
      int rlA = r0 + rr + hi * 8;
      int dnA = dstl[rlA], dnB = dstl[rlA + 16];
      atomicAdd(&agg[dnA * DIM + 0 + m], a0[rr]);
      atomicAdd(&agg[dnA * DIM + 16 + m], a1[rr]);
      atomicAdd(&agg[dnA * DIM + 32 + m], a2[rr]);
      atomicAdd(&agg[dnA * DIM + 48 + m], a3[rr]);
      atomicAdd(&agg[dnB * DIM + 0 + m], b0[rr]);
      atomicAdd(&agg[dnB * DIM + 16 + m], b1[rr]);
      atomicAdd(&agg[dnB * DIM + 32 + m], b2[rr]);
      atomicAdd(&agg[dnB * DIM + 48 + m], b3[rr]);
    }
  } else {
#pragma unroll
    for (int rr = 0; rr < 8; ++rr) {
      int rlA = r0 + rr + hi * 8;
      if (eb + rlA < N_EDGES) {
        int dn = dstl[rlA];
        atomicAdd(&agg[dn * DIM + 0 + m], a0[rr]);
        atomicAdd(&agg[dn * DIM + 16 + m], a1[rr]);
        atomicAdd(&agg[dn * DIM + 32 + m], a2[rr]);
        atomicAdd(&agg[dn * DIM + 48 + m], a3[rr]);
      }
      if (eb + rlA + 16 < N_EDGES) {
        int dn = dstl[rlA + 16];
        atomicAdd(&agg[dn * DIM + 0 + m], b0[rr]);
        atomicAdd(&agg[dn * DIM + 16 + m], b1[rr]);
        atomicAdd(&agg[dn * DIM + 32 + m], b2[rr]);
        atomicAdd(&agg[dn * DIM + 48 + m], b3[rr]);
      }
    }
  }
}

// ---------------- combine + GRU cell (out == h) ----------------
__global__ void combine_gru(const float* __restrict__ agg, const float* __restrict__ invd,
                            const float* __restrict__ root, const float* __restrict__ cbias,
                            const float* __restrict__ wih, const float* __restrict__ whh,
                            const float* __restrict__ bih, const float* __restrict__ bhh,
                            float* __restrict__ out, _Float16* __restrict__ xs) {
  __shared__ float ov[4][DIM];
  __shared__ float mv[4][DIM];
  int ln = threadIdx.x / DIM, d = threadIdx.x % DIM;
  int n = blockIdx.x * 4 + ln;  // N divisible by 4
  ov[ln][d] = out[n * DIM + d];
  __syncthreads();
  float s = agg[n * DIM + d] * invd[n] + cbias[d];
  for (int j = 0; j < DIM; ++j) s += ov[ln][j] * root[j * DIM + d];
  mv[ln][d] = fmaxf(s, 0.f);
  __syncthreads();
  float xr = bih[d], xz = bih[DIM + d], xn = bih[2 * DIM + d];
  float hr = bhh[d], hz = bhh[DIM + d], hn = bhh[2 * DIM + d];
  for (int j = 0; j < DIM; ++j) {
    float mj = mv[ln][j], hj = ov[ln][j];
    xr += mj * wih[(0 * DIM + d) * DIM + j];
    xz += mj * wih[(1 * DIM + d) * DIM + j];
    xn += mj * wih[(2 * DIM + d) * DIM + j];
    hr += hj * whh[(0 * DIM + d) * DIM + j];
    hz += hj * whh[(1 * DIM + d) * DIM + j];
    hn += hj * whh[(2 * DIM + d) * DIM + j];
  }
  float rg = sigm(xr + hr), zg = sigm(xz + hz);
  float ng = tanhf(xn + rg * hn);
  float hnew = (1.f - zg) * ng + zg * ov[ln][d];
  out[n * DIM + d] = hnew;
  xs[n * DIM + d] = (_Float16)hnew;
}

// ---------------- Set2Set ----------------
__global__ void s2s_lstm(const float* __restrict__ qstar, const float* __restrict__ wih,
                         const float* __restrict__ whh, const float* __restrict__ bih,
                         const float* __restrict__ bhh, float* __restrict__ hs,
                         float* __restrict__ cs) {
  __shared__ float hsl[4][DIM];
  int t = blockIdx.x * 256 + threadIdx.x;
  int b = t / DIM, d = t % DIM, lb = threadIdx.x / DIM;
  hsl[lb][d] = hs[b * DIM + d];
  __syncthreads();
  float g[4];
  const float* qs = qstar + b * 2 * DIM;
#pragma unroll
  for (int gi = 0; gi < 4; ++gi) {
    int k = gi * DIM + d;
    float s = bih[k] + bhh[k];
    const float* wi = wih + k * 2 * DIM;
    for (int j = 0; j < 2 * DIM; ++j) s += qs[j] * wi[j];
    const float* wh = whh + k * DIM;
    for (int j = 0; j < DIM; ++j) s += hsl[lb][j] * wh[j];
    g[gi] = s;
  }
  float c = sigm(g[1]) * cs[b * DIM + d] + sigm(g[0]) * tanhf(g[2]);
  cs[b * DIM + d] = c;
  hs[b * DIM + d] = sigm(g[3]) * tanhf(c);
}

__global__ void s2s_att(const float* __restrict__ out, const float* __restrict__ hs,
                        const int* __restrict__ batch, float* __restrict__ e_ws,
                        unsigned* __restrict__ emk) {
  int n = blockIdx.x * blockDim.x + threadIdx.x;
  if (n >= N_NODES) return;
  int b = batch[n];
  const float* orow = out + n * DIM;
  const float* q = hs + b * DIM;
  float s = 0.f;
  for (int j = 0; j < DIM; ++j) s += orow[j] * q[j];
  e_ws[n] = s;
  atomicMax(&emk[b], fkey(s));
}

__global__ void s2s_ee(const float* __restrict__ e_ws, const int* __restrict__ batch,
                       const unsigned* __restrict__ emk, float* __restrict__ ee_ws,
                       float* __restrict__ denom) {
  int n = blockIdx.x * blockDim.x + threadIdx.x;
  if (n >= N_NODES) return;
  int b = batch[n];
  float ev = __expf(e_ws[n] - keyf(emk[b]));
  ee_ws[n] = ev;
  atomicAdd(&denom[b], ev);
}

__global__ void s2s_rvec(const float* __restrict__ ee_ws, const float* __restrict__ denom,
                         const int* __restrict__ batch, const float* __restrict__ out,
                         float* __restrict__ rvec) {
  int t = blockIdx.x * blockDim.x + threadIdx.x;
  if (t >= N_NODES * DIM) return;
  int n = t / DIM, d = t % DIM;
  int b = batch[n];
  float a = ee_ws[n] / fmaxf(denom[b], 1e-16f);
  atomicAdd(&rvec[b * DIM + d], a * out[t]);
}

__global__ void s2s_qstar(const float* __restrict__ hs, const float* __restrict__ rvec,
                          float* __restrict__ qstar) {
  int t = blockIdx.x * blockDim.x + threadIdx.x;
  if (t >= NB * 2 * DIM) return;
  int b = t / (2 * DIM), d = t % (2 * DIM);
  qstar[t] = (d < DIM) ? hs[b * DIM + d] : rvec[b * DIM + d - DIM];
}

// ---------------- readout ----------------
__global__ void final_kernel(const float* __restrict__ qstar, const float* __restrict__ w1,
                             const float* __restrict__ b1, const float* __restrict__ w2,
                             const float* __restrict__ b2, float* __restrict__ y) {
  __shared__ float part[DIM];
  int b = blockIdx.x, d = threadIdx.x;
  const float* qs = qstar + b * 2 * DIM;
  float s = b1[d];
  for (int j = 0; j < 2 * DIM; ++j) s += qs[j] * w1[d * 2 * DIM + j];
  part[d] = fmaxf(s, 0.f) * w2[d];
  __syncthreads();
  for (int off = 32; off > 0; off >>= 1) {
    if (d < off) part[d] += part[d + off];
    __syncthreads();
  }
  if (d == 0) y[b] = part[0] + b2[0];
}

// ---------------- launch ----------------
extern "C" void kernel_launch(void* const* d_in, const int* in_sizes, int n_in,
                              void* d_out, int out_size, void* d_ws, size_t ws_size,
                              hipStream_t stream) {
  const float* x = (const float*)d_in[0];
  const float* edge_attr = (const float*)d_in[1];
  const float* lin0_w = (const float*)d_in[2];
  const float* lin0_b = (const float*)d_in[3];
  const float* nn1_w = (const float*)d_in[4];
  const float* nn1_b = (const float*)d_in[5];
  const float* nn2_w = (const float*)d_in[6];
  const float* nn2_b = (const float*)d_in[7];
  const float* root = (const float*)d_in[8];
  const float* conv_bias = (const float*)d_in[9];
  const float* gru_w_ih = (const float*)d_in[10];
  const float* gru_w_hh = (const float*)d_in[11];
  const float* gru_b_ih = (const float*)d_in[12];
  const float* gru_b_hh = (const float*)d_in[13];
  const float* lstm_w_ih = (const float*)d_in[14];
  const float* lstm_w_hh = (const float*)d_in[15];
  const float* lstm_b_ih = (const float*)d_in[16];
  const float* lstm_b_hh = (const float*)d_in[17];
  const float* lin1_w = (const float*)d_in[18];
  const float* lin1_b = (const float*)d_in[19];
  const float* lin2_w = (const float*)d_in[20];
  const float* lin2_b = (const float*)d_in[21];
  const int* edge_index = (const int*)d_in[22];
  const int* batch = (const int*)d_in[23];
  float* y = (float*)d_out;

  char* ws = (char*)d_ws;
  size_t off = 0;
  auto alloc = [&](size_t bytes) -> void* {
    off = (off + 255) & ~(size_t)255;
    void* p = ws + off;
    off += bytes;
    return p;
  };
  float* out_f = (float*)alloc((size_t)N_NODES * DIM * 4);
  _Float16* xs_h = (_Float16*)alloc((size_t)N_NODES * DIM * 2);
  _Float16* he_h = (_Float16*)alloc((size_t)N_EDGES * HH * 2);
  _Float16* w2t = (_Float16*)alloc((size_t)64 * 8192 * 2);
  _Float16* bbt = (_Float16*)alloc((size_t)DIM * DIM * 2);
  _Float16* n1bt = (_Float16*)alloc((size_t)HH * 64 * 2);
  float* agg = (float*)alloc((size_t)N_NODES * DIM * 4);
  float* deg = (float*)alloc((size_t)N_NODES * 4);
  float* invd = (float*)alloc((size_t)N_NODES * 4);
  float* qstar = (float*)alloc((size_t)NB * 2 * DIM * 4);
  float* hs = (float*)alloc((size_t)NB * DIM * 4);
  float* cs = (float*)alloc((size_t)NB * DIM * 4);
  float* e_ws = (float*)alloc((size_t)N_NODES * 4);
  float* ee_ws = (float*)alloc((size_t)N_NODES * 4);
  unsigned* emk = (unsigned*)alloc((size_t)NB * 4);
  float* denom = (float*)alloc((size_t)NB * 4);
  float* rvec = (float*)alloc((size_t)NB * DIM * 4);
  (void)in_sizes; (void)n_in; (void)out_size; (void)ws_size;

  const int EB = (N_EDGES + EPB - 1) / EPB;  // 782

  prep_w2t<<<(64 * 8192 + 255) / 256, 256, 0, stream>>>(nn2_w, w2t);
  prep_bbt<<<(DIM * DIM + 255) / 256, 256, 0, stream>>>(nn2_b, bbt);
  prep_n1bt<<<(HH * 64 + 255) / 256, 256, 0, stream>>>(nn1_w, n1bt);

  lin0_kernel<<<N_NODES / 4, 256, 0, stream>>>(x, lin0_w, lin0_b, out_f, xs_h);
  nn1_wmma<<<EB, 256, 0, stream>>>(edge_attr, nn1_b, n1bt, he_h);

  zero_f<<<(N_NODES + 255) / 256, 256, 0, stream>>>(deg, N_NODES);
  deg_kernel<<<(N_EDGES + 255) / 256, 256, 0, stream>>>(edge_index, deg);
  invdeg_kernel<<<(N_NODES + 255) / 256, 256, 0, stream>>>(deg, invd);

  for (int step = 0; step < 2; ++step) {
    zero_f<<<(N_NODES * DIM + 255) / 256, 256, 0, stream>>>(agg, N_NODES * DIM);
    conv_msg<<<EB, 128, 0, stream>>>(edge_index, xs_h, he_h, w2t, bbt, agg);
    combine_gru<<<N_NODES / 4, 256, 0, stream>>>(agg, invd, root, conv_bias, gru_w_ih,
                                                 gru_w_hh, gru_b_ih, gru_b_hh, out_f, xs_h);
  }

  zero_f<<<(NB * 2 * DIM + 255) / 256, 256, 0, stream>>>(qstar, NB * 2 * DIM);
  zero_f<<<(NB * DIM + 255) / 256, 256, 0, stream>>>(hs, NB * DIM);
  zero_f<<<(NB * DIM + 255) / 256, 256, 0, stream>>>(cs, NB * DIM);

  for (int step = 0; step < 2; ++step) {
    zero_u<<<1, 256, 0, stream>>>(emk, NB);
    zero_f<<<1, 256, 0, stream>>>(denom, NB);
    zero_f<<<(NB * DIM + 255) / 256, 256, 0, stream>>>(rvec, NB * DIM);
    s2s_lstm<<<NB * DIM / 256, 256, 0, stream>>>(qstar, lstm_w_ih, lstm_w_hh, lstm_b_ih,
                                                 lstm_b_hh, hs, cs);
    s2s_att<<<(N_NODES + 255) / 256, 256, 0, stream>>>(out_f, hs, batch, e_ws, emk);
    s2s_ee<<<(N_NODES + 255) / 256, 256, 0, stream>>>(e_ws, batch, emk, ee_ws, denom);
    s2s_rvec<<<(N_NODES * DIM + 255) / 256, 256, 0, stream>>>(ee_ws, denom, batch, out_f,
                                                              rvec);
    s2s_qstar<<<(NB * 2 * DIM + 255) / 256, 256, 0, stream>>>(hs, rvec, qstar);
  }

  final_kernel<<<NB, DIM, 0, stream>>>(qstar, lin1_w, lin1_b, lin2_w, lin2_b, y);
}
